// SelfAttention_36567351558519
// MI455X (gfx1250) — matrix operand order repaired and checked by reading, not compiled, not run on hardware
//
#include <hip/hip_runtime.h>
#include <hip/hip_bf16.h>

typedef __attribute__((ext_vector_type(16))) _Float16 v16h;
typedef __attribute__((ext_vector_type(8)))  _Float16 v8h;
typedef __attribute__((ext_vector_type(8)))  float    v8f;
typedef __attribute__((ext_vector_type(4)))  unsigned int u32x4;
typedef __attribute__((ext_vector_type(8)))  int          i32x8;
typedef __attribute__((ext_vector_type(4)))  int          i32x4;

#define B_   4
#define C_   512
#define C8_  64
#define N_   4096
#define EPS_ 1e-5f
#define SHLD 66   // padded LDS stride for S tile (bank-conflict spread)

#if defined(__has_builtin)
#if __has_builtin(__builtin_amdgcn_tensor_load_to_lds) && __has_builtin(__builtin_amdgcn_s_wait_tensorcnt)
#define HAVE_TDM 1
#endif
#endif
#ifndef HAVE_TDM
#define HAVE_TDM 0
#endif

// ---------------------------------------------------------------------------
// WMMA helpers (CDNA5 16x16x32 f16 -> f32)
// ---------------------------------------------------------------------------
__device__ __forceinline__ v8f wmma_f16(v16h a, v16h b, v8f c) {
    return __builtin_amdgcn_wmma_f32_16x16x32_f16(false, a, false, b,
                                                  (short)0, c, false, false);
}

// A fragment: A is MxK row-major. row = lane&15; K = (h/8)*16 + (lane>>4)*8 + h%8
__device__ __forceinline__ v16h frag_a(const _Float16* Abase, int lda, int lane, int k0) {
    const _Float16* p = Abase + (long)(lane & 15) * lda + k0 + ((lane >> 4) << 3);
    union { v16h v; v8h h[2]; } u;
    u.h[0] = *(const v8h*)(p);
    u.h[1] = *(const v8h*)(p + 16);
    return u.v;
}

// B fragment from B^T row-major (N x K). col = lane&15; K = (h/8)*16 + (lane>>4)*8 + h%8
__device__ __forceinline__ v16h frag_bt(const _Float16* BTbase, int ldbt, int lane, int k0) {
    const _Float16* p = BTbase + (long)(lane & 15) * ldbt + k0 + ((lane >> 4) << 3);
    union { v16h v; v8h h[2]; } u;
    u.h[0] = *(const v8h*)(p);
    u.h[1] = *(const v8h*)(p + 16);
    return u.v;
}

#if HAVE_TDM
// ---------------------------------------------------------------------------
// TDM: DMA a 2-D tile (C_ rows x 64 cols, f16, row stride N_) from global
// into LDS (row-major, ld=64). Descriptor per CDNA5 ISA 08_async_tensor §8.
// Issued wave-uniform (EXEC ignored by tensor ops); tracked by TENSORcnt.
// ---------------------------------------------------------------------------
__device__ __forceinline__ void tdm_load_v_tile(const _Float16* gsrc, void* lds_dst) {
    unsigned long long ga = (unsigned long long)(uintptr_t)gsrc;   // byte addr
    unsigned lds = (unsigned)(uintptr_t)lds_dst;                   // LDS offset (addr[31:0])
    u32x4 g0;
    g0[0] = 1u;                                      // count=1, user mode, no gather
    g0[1] = lds;                                     // lds_addr
    g0[2] = (unsigned)(ga & 0xffffffffull);          // global_addr[31:0]
    g0[3] = (unsigned)((ga >> 32) & 0x1ffffffull)    // global_addr[56:32]
          | (2u << 30);                              // type = 2 ("image")
    i32x8 g1;
    g1[0] = (1 << 16);                               // wg_mask=0, data_size=1 (2B)
    g1[1] = (int)((unsigned)(N_ & 0xffff) << 16);    // tensor_dim0[15:0]
    g1[2] = (int)(((unsigned)N_ >> 16) & 0xffff)     // tensor_dim0[31:16]
          | (int)((unsigned)(C_ & 0xffff) << 16);    // tensor_dim1[15:0]
    g1[3] = (int)(((unsigned)C_ >> 16) & 0xffff)     // tensor_dim1[31:16]
          | (64 << 16);                              // tile_dim0 = 64
    g1[4] = (int)(C_ & 0xffff);                      // tile_dim1 = C_, tile_dim2 = 0
    g1[5] = N_;                                      // tensor_dim0_stride[31:0]
    g1[6] = 0;                                       // stride0 hi, stride1 lo
    g1[7] = 0;
    i32x4 z4 = {0, 0, 0, 0};
#if __clang_major__ >= 23
    i32x8 z8 = {0, 0, 0, 0, 0, 0, 0, 0};
    __builtin_amdgcn_tensor_load_to_lds(g0, g1, z4, z4, z8, 0);
#else
    __builtin_amdgcn_tensor_load_to_lds(g0, g1, z4, z4, 0);
#endif
}
#endif

// ---------------------------------------------------------------------------
// Kernel 0: x (B,C,N) f32 -> XT (B,N,C) f16, LDS-tiled transpose
// ---------------------------------------------------------------------------
__global__ void k_transpose_cvt(const float* __restrict__ x, _Float16* __restrict__ XT) {
    __shared__ float tile[32][33];
    const int b  = blockIdx.z;
    const int c0 = blockIdx.y * 32;
    const int i0 = blockIdx.x * 32;
    const int tx = threadIdx.x, ty = threadIdx.y; // (32, 8)
    #pragma unroll
    for (int r = ty; r < 32; r += 8)
        tile[r][tx] = x[((long)b * C_ + c0 + r) * N_ + i0 + tx];
    __syncthreads();
    #pragma unroll
    for (int r = ty; r < 32; r += 8)
        XT[((long)b * N_ + i0 + r) * C_ + c0 + tx] = (_Float16)tile[tx][r];
}

// ---------------------------------------------------------------------------
// Kernel 1: f32 -> f16 weight conversion
// ---------------------------------------------------------------------------
__global__ void k_cvt_f16(const float* __restrict__ src, _Float16* __restrict__ dst, int n) {
    int i = blockIdx.x * blockDim.x + threadIdx.x;
    if (i < n) dst[i] = (_Float16)src[i];
}

// ---------------------------------------------------------------------------
// Kernel 2: generic batched tile-GEMM, one 16x16 D tile per wave.
// ---------------------------------------------------------------------------
__global__ __launch_bounds__(256)
void k_gemm16(const _Float16* __restrict__ A,  long sAb, int lda,
              const _Float16* __restrict__ BT, long sBb, int ldbt,
              _Float16* __restrict__ D,        long sDb, int ldd,
              const float* __restrict__ bias, int biasPerRow,
              int Mtiles, int Ntiles, int Ksteps) {
    const int lane = threadIdx.x & 31;
    const int wave = threadIdx.x >> 5;
    const int b    = blockIdx.y;
    const int tile = blockIdx.x * 8 + wave;
    if (tile >= Mtiles * Ntiles) return;   // wave-uniform
    const int tm = tile / Ntiles;
    const int tn = tile % Ntiles;

    const _Float16* Ab = A  + (long)b * sAb + (long)tm * 16 * lda;
    const _Float16* Bb = BT + (long)b * sBb + (long)tn * 16 * ldbt;

    v8f acc = {};
    for (int ks = 0; ks < Ksteps; ++ks) {
        v16h af = frag_a (Ab, lda,  lane, ks * 32);
        v16h bf = frag_bt(Bb, ldbt, lane, ks * 32);
        acc = wmma_f16(af, bf, acc);
    }

    const int col    = lane & 15;
    const int rowoff = (lane >> 4) << 3;
    const int colg   = tn * 16 + col;
    _Float16* Dout   = D + (long)b * sDb;
    #pragma unroll
    for (int r = 0; r < 8; ++r) {
        int rowg = tm * 16 + r + rowoff;
        float bsv = biasPerRow ? bias[rowg] : bias[colg];
        Dout[(long)rowg * ldd + colg] = (_Float16)(acc[r] + bsv);
    }
}

// ---------------------------------------------------------------------------
// Kernel 3: fused flash attention with TDM-staged V tiles (double-buffered).
// Block = 256 thr (8 waves) = 64 query rows. Wave w owns channels [64w,64w+64).
// ---------------------------------------------------------------------------
__global__ __launch_bounds__(256)
void k_flash_attn(const _Float16* __restrict__ QT,   // (B,N,64)
                  const _Float16* __restrict__ KT,   // (B,N,64)
                  const _Float16* __restrict__ Vc,   // (B,C,N)
                  const float* __restrict__ x,       // (B,C,N)
                  const float* __restrict__ gamma,
                  float* __restrict__ preout) {      // (B,C,N)
    __shared__ float    Sh[64 * SHLD];
    __shared__ _Float16 Ph[64 * 64];
    __shared__ _Float16 Vs[2][C_ * 64];   // staged V block: C_ rows x 64 j, ld=64
    __shared__ float    mS[64], lS[64], aS[64];

    const int tid  = threadIdx.x;
    const int lane = tid & 31;
    const int wave = tid >> 5;
    const int b    = blockIdx.x >> 6;          // N_/64 row-blocks per batch
    const int i0   = (blockIdx.x & 63) << 6;

    const _Float16* Qb = QT + (long)b * N_ * C8_;
    const _Float16* Kb = KT + (long)b * N_ * C8_;
    const _Float16* Vb = Vc + (long)b * C_ * N_;

    if (tid < 64) { mS[tid] = -3.0e38f; lS[tid] = 0.f; }

    v8f o[16];
    #pragma unroll
    for (int t = 0; t < 16; ++t) o[t] = v8f{};

    const int col    = lane & 15;
    const int rowoff = (lane >> 4) << 3;

    int buf = 0;
#if HAVE_TDM
    if (wave == 0) tdm_load_v_tile(Vb, &Vs[0][0]);   // prologue: block j0=0
#endif
    __syncthreads();

    for (int j0 = 0; j0 < N_; j0 += 64) {
#if HAVE_TDM
        if (wave == 0 && j0 + 64 < N_)               // async-prefetch next block
            tdm_load_v_tile(Vb + j0 + 64, &Vs[buf ^ 1][0]);
#endif
        // ---- S tiles: wave computes tiles 2w and 2w+1 of the 4x4 grid -----
        #pragma unroll
        for (int tt = 0; tt < 2; ++tt) {
            const int t  = wave * 2 + tt;
            const int sr = t >> 2, sc = t & 3;
            v8f s = {};
            #pragma unroll
            for (int ks = 0; ks < 2; ++ks) {
                v16h af = frag_a (Qb + (long)(i0 + sr * 16) * C8_, C8_, lane, ks * 32);
                v16h bf = frag_bt(Kb + (long)(j0 + sc * 16) * C8_, C8_, lane, ks * 32);
                s = wmma_f16(af, bf, s);
            }
            #pragma unroll
            for (int r = 0; r < 8; ++r)
                Sh[(sr * 16 + r + rowoff) * SHLD + sc * 16 + col] = s[r];
        }
        if (j0 + 64 < N_)
            __builtin_prefetch(Kb + (long)(j0 + 64 + wave * 8) * C8_, 0, 0);
        __syncthreads();

        // ---- online softmax: quad of lanes per row (all 256 threads) ------
        {
            const int row = tid >> 2;
            const int seg = tid & 3;
            const float* srow = &Sh[row * SHLD + seg * 16];
            float mloc = -3.0e38f;
            #pragma unroll
            for (int j = 0; j < 16; ++j) mloc = fmaxf(mloc, srow[j]);
            mloc = fmaxf(mloc, __shfl_xor(mloc, 1));
            mloc = fmaxf(mloc, __shfl_xor(mloc, 2));
            float mOld = mS[row];
            float mNew = fmaxf(mOld, mloc);
            float sum = 0.f;
            _Float16* prow = &Ph[row * 64 + seg * 16];
            #pragma unroll
            for (int j = 0; j < 16; ++j) {
                float p = __expf(srow[j] - mNew);
                sum += p;
                prow[j] = (_Float16)p;
            }
            sum += __shfl_xor(sum, 1);
            sum += __shfl_xor(sum, 2);
            if (seg == 0) {
                float alpha = __expf(mOld - mNew);
                mS[row] = mNew;
                lS[row] = lS[row] * alpha + sum;
                aS[row] = alpha;
            }
        }

#if HAVE_TDM
        if (wave == 0) {      // retire the DMA for the current block
            if (j0 + 64 < N_) __builtin_amdgcn_s_wait_tensorcnt(1);
            else              __builtin_amdgcn_s_wait_tensorcnt(0);
        }
#else
        // fallback: cooperative copy of current V block into LDS
        for (int r = tid; r < C_; r += 256) {
            const v8h* src = (const v8h*)(Vb + (long)r * N_ + j0);
            v8h* dst = (v8h*)&Vs[buf][r * 64];
            dst[0] = src[0];
            dst[1] = src[1];
            dst[2] = src[2];
            dst[3] = src[3];
        }
#endif
        __syncthreads();   // softmax results + V tile visible to all waves

        // ---- rescale accumulators by alpha[row] ---------------------------
        #pragma unroll
        for (int tr = 0; tr < 4; ++tr) {
            float al[8];
            #pragma unroll
            for (int r = 0; r < 8; ++r) al[r] = aS[tr * 16 + r + rowoff];
            #pragma unroll
            for (int cc = 0; cc < 4; ++cc)
                #pragma unroll
                for (int r = 0; r < 8; ++r) o[tr * 4 + cc][r] *= al[r];
        }

        // ---- O += P @ V^T for this wave's 64-channel slice (all from LDS) -
        #pragma unroll
        for (int ks = 0; ks < 2; ++ks) {
            v16h bf[4];
            #pragma unroll
            for (int cc = 0; cc < 4; ++cc)
                bf[cc] = frag_bt((const _Float16*)&Vs[buf][(wave * 64 + cc * 16) * 64],
                                 64, lane, ks * 32);
            #pragma unroll
            for (int tr = 0; tr < 4; ++tr) {
                v16h af = frag_a((const _Float16*)&Ph[tr * 16 * 64], 64, lane, ks * 32);
                #pragma unroll
                for (int cc = 0; cc < 4; ++cc)
                    o[tr * 4 + cc] = wmma_f16(af, bf[cc], o[tr * 4 + cc]);
            }
        }
        __syncthreads();   // protect Sh/Ph/Vs[buf] before reuse
        buf ^= 1;
    }

    // ---- epilogue: /l, residual + gamma, write preout ---------------------
    const float g = *gamma;
    #pragma unroll
    for (int tr = 0; tr < 4; ++tr) {
        float linv[8];
        #pragma unroll
        for (int r = 0; r < 8; ++r) linv[r] = 1.0f / lS[tr * 16 + r + rowoff];
        #pragma unroll
        for (int cc = 0; cc < 4; ++cc) {
            const int c = wave * 64 + cc * 16 + col;
            #pragma unroll
            for (int r = 0; r < 8; ++r) {
                const int i = i0 + tr * 16 + r + rowoff;
                const long idx = ((long)b * C_ + c) * N_ + i;
                preout[idx] = g * (o[tr * 4 + cc][r] * linv[r]) + x[idx];
            }
        }
    }
}

// ---------------------------------------------------------------------------
// Kernel 4: per-channel batch statistics (mean, inv-std)
// ---------------------------------------------------------------------------
__global__ __launch_bounds__(256)
void k_bn_stats(const float* __restrict__ pre, float* __restrict__ stats) {
    __shared__ float rs[256], rs2[256];
    const int c   = blockIdx.x;
    const int tid = threadIdx.x;
    float s = 0.f, s2 = 0.f;
    for (int idx = tid; idx < B_ * N_; idx += 256) {
        const int b = idx >> 12;
        const int i = idx & (N_ - 1);
        float v = pre[((long)b * C_ + c) * N_ + i];
        s += v; s2 += v * v;
    }
    rs[tid] = s; rs2[tid] = s2;
    __syncthreads();
    for (int w = 128; w > 0; w >>= 1) {
        if (tid < w) { rs[tid] += rs[tid + w]; rs2[tid] += rs2[tid + w]; }
        __syncthreads();
    }
    if (tid == 0) {
        const float inv_n = 1.0f / (float)(B_ * N_);
        float mean = rs[0] * inv_n;
        float var  = rs2[0] * inv_n - mean * mean;
        stats[c]      = mean;
        stats[C_ + c] = rsqrtf(var + EPS_);
    }
}

// ---------------------------------------------------------------------------
// Kernel 5: apply BatchNorm affine
// ---------------------------------------------------------------------------
__global__ void k_bn_apply(const float* __restrict__ pre, const float* __restrict__ stats,
                           const float* __restrict__ bn_w, const float* __restrict__ bn_b,
                           float* __restrict__ out) {
    const long idx = (long)blockIdx.x * blockDim.x + threadIdx.x;
    if (idx >= (long)B_ * C_ * N_) return;
    const int c = (int)((idx >> 12) & (C_ - 1));
    out[idx] = (pre[idx] - stats[c]) * stats[C_ + c] * bn_w[c] + bn_b[c];
}

// ---------------------------------------------------------------------------
extern "C" void kernel_launch(void* const* d_in, const int* in_sizes, int n_in,
                              void* d_out, int out_size, void* d_ws, size_t ws_size,
                              hipStream_t stream) {
    const float* x     = (const float*)d_in[0];
    const float* Wq    = (const float*)d_in[1];
    const float* bq    = (const float*)d_in[2];
    const float* Wk    = (const float*)d_in[3];
    const float* bk    = (const float*)d_in[4];
    const float* Wv    = (const float*)d_in[5];
    const float* bv    = (const float*)d_in[6];
    const float* gamma = (const float*)d_in[7];
    const float* bn_w  = (const float*)d_in[8];
    const float* bn_b  = (const float*)d_in[9];
    float* out = (float*)d_out;

    char* ws = (char*)d_ws;
    size_t off = 0;
    auto alloc = [&](size_t bytes) -> void* {
        off = (off + 255) & ~(size_t)255;
        void* p = ws + off;
        off += bytes;
        return p;
    };

    _Float16* XT   = (_Float16*)alloc((size_t)B_ * N_ * C_ * 2);
    _Float16* Wq16 = (_Float16*)alloc((size_t)C8_ * C_ * 2);
    _Float16* Wk16 = (_Float16*)alloc((size_t)C8_ * C_ * 2);
    _Float16* Wv16 = (_Float16*)alloc((size_t)C_ * C_ * 2);
    _Float16* QTb  = (_Float16*)alloc((size_t)B_ * N_ * C8_ * 2);
    _Float16* KTb  = (_Float16*)alloc((size_t)B_ * N_ * C8_ * 2);
    _Float16* Vcb  = (_Float16*)alloc((size_t)B_ * C_ * N_ * 2);
    float*    pre  = (float*)   alloc((size_t)B_ * C_ * N_ * 4);
    float*    stat = (float*)   alloc((size_t)2 * C_ * 4);

    k_transpose_cvt<<<dim3(N_ / 32, C_ / 32, B_), dim3(32, 8), 0, stream>>>(x, XT);

    k_cvt_f16<<<(C8_ * C_ + 255) / 256, 256, 0, stream>>>(Wq, Wq16, C8_ * C_);
    k_cvt_f16<<<(C8_ * C_ + 255) / 256, 256, 0, stream>>>(Wk, Wk16, C8_ * C_);
    k_cvt_f16<<<(C_ * C_  + 255) / 256, 256, 0, stream>>>(Wv, Wv16, C_ * C_);

    k_gemm16<<<dim3(128, B_), 256, 0, stream>>>(
        XT, (long)N_ * C_, C_,       Wq16, 0, C_,
        QTb, (long)N_ * C8_, C8_,    bq, 0, N_ / 16, C8_ / 16, C_ / 32);
    k_gemm16<<<dim3(128, B_), 256, 0, stream>>>(
        XT, (long)N_ * C_, C_,       Wk16, 0, C_,
        KTb, (long)N_ * C8_, C8_,    bk, 0, N_ / 16, C8_ / 16, C_ / 32);
    k_gemm16<<<dim3(1024, B_), 256, 0, stream>>>(
        Wv16, 0, C_,                 XT, (long)N_ * C_, C_,
        Vcb, (long)C_ * N_, N_,      bv, 1, C_ / 16, N_ / 16, C_ / 32);

    k_flash_attn<<<B_ * (N_ / 64), 256, 0, stream>>>(QTb, KTb, Vcb, x, gamma, pre);

    k_bn_stats<<<C_, 256, 0, stream>>>(pre, stat);
    k_bn_apply<<<(int)(((long)B_ * C_ * N_ + 255) / 256), 256, 0, stream>>>(
        pre, stat, bn_w, bn_b, out);
}
